// DualImageAttnProcessor_40518721470713
// MI455X (gfx1250) — matrix operand order repaired and checked by reading, not compiled, not run on hardware
//
#include <hip/hip_runtime.h>
#include <hip/hip_bf16.h>
#include <stdint.h>

// ---------------------------------------------------------------------------
// MI455X (gfx1250) implementation.
// - bf16 WMMA (v_wmma_f32_16x16x32_bf16) with f32 accumulation everywhere.
// - Wo folding: (base+id+hair)@Wo + 3*bo  (SCALE_ID=SCALE_HAIR=1).
// - GEMMs: 128x128 block tile, 8 waves x (32x64), double-buffered LDS with
//   GLOBAL_LOAD_ASYNC_TO_LDS_B128 staging (ASYNCcnt + s_wait_asynccnt).
// - Weights pre-transposed to bf16 [N,K] so B staging is a pure async copy.
// ---------------------------------------------------------------------------

typedef __attribute__((ext_vector_type(16))) __bf16 v16bf;
typedef __attribute__((ext_vector_type(8)))  float  v8f;
typedef int v4i __attribute__((vector_size(16)));   // builtin's pointee type

union FragAB {
    uint32_t u[8];
    uint16_t h[16];
    v16bf    v;
};

__device__ __forceinline__ uint16_t f32_to_bf16(float f) {
    uint32_t u = __float_as_uint(f);
    uint32_t r = u + 0x7FFFu + ((u >> 16) & 1u);   // round-to-nearest-even
    return (uint16_t)(r >> 16);
}

__device__ __forceinline__ v8f zero8() {
    v8f z;
#pragma unroll
    for (int i = 0; i < 8; ++i) z[i] = 0.0f;
    return z;
}

// --------------------------- async global->LDS ------------------------------
#if defined(__gfx1250__) && __has_builtin(__builtin_amdgcn_global_load_async_to_lds_b128)
#define HAVE_ASYNC_LDS 1
#else
#define HAVE_ASYNC_LDS 0
#endif

__device__ __forceinline__ void async_cp16(const void* g, void* l) {
#if HAVE_ASYNC_LDS
    __builtin_amdgcn_global_load_async_to_lds_b128(
        (__attribute__((address_space(1))) v4i*)(void*)(g),
        (__attribute__((address_space(3))) v4i*)(l), 0, 0);
#else
    *(uint4*)l = *(const uint4*)g;
#endif
}

__device__ __forceinline__ void wait_async_lds() {
#if HAVE_ASYNC_LDS
    asm volatile("s_wait_asynccnt 0" ::: "memory");
#endif
}

// ---------------------------------------------------------------------------
// Weight transpose + convert:  Wt[n,k] (bf16) = W[k,n] (f32)
// ---------------------------------------------------------------------------
__global__ __launch_bounds__(256) void transpose_bf16(
    const float* __restrict__ W, uint16_t* __restrict__ Wt, int K, int N)
{
    __shared__ float tile[32][33];
    const int nb = blockIdx.x * 32;
    const int kb = blockIdx.y * 32;
    const int tx = threadIdx.x & 31;
    const int ty = threadIdx.x >> 5;          // 0..7
#pragma unroll
    for (int i = 0; i < 32; i += 8)
        tile[ty + i][tx] = W[(size_t)(kb + ty + i) * N + nb + tx];
    __syncthreads();
#pragma unroll
    for (int i = 0; i < 32; i += 8)
        Wt[(size_t)(nb + ty + i) * K + kb + tx] = f32_to_bf16(tile[tx][ty + i]);
}

// ---------------------------------------------------------------------------
// WMMA GEMM:  C[M,N] = A[M,K] * Bt[N,K]^T (+ bias_scale*bias)
//   A: f32 (VALU-converted during staging) or bf16 (async-copied).
//   Bt: bf16, pre-transposed [N,K] -> B staging is a pure async copy.
// Block tile 128(M) x 128(N), K-step 32, double-buffered LDS.
// 8 waves: 4 along M x 2 along N; each wave computes 2x4 WMMA tiles.
// LDS row stride 40 halves => conflict-free fragment ds_loads.
// ---------------------------------------------------------------------------
template<bool A_IS_F32, bool OUT_BF16>
__global__ __launch_bounds__(256) void gemm_wmma(
    const void* __restrict__ Aptr, const uint16_t* __restrict__ Bt,
    void* __restrict__ Cptr, const float* __restrict__ bias, float bias_scale,
    int M, int N, int K)
{
    __shared__ __align__(16) uint16_t As[2][128][40];   // [buf][m][k]
    __shared__ __align__(16) uint16_t Bs[2][128][40];   // [buf][n][k]

    const int tid  = threadIdx.x;
    const int lane = tid & 31;
    const int w    = tid >> 5;
    const int wm   = w & 3;             // 4 waves along M
    const int wn   = w >> 2;            // 2 waves along N
    const int m    = lane & 15;
    const int hf   = lane >> 4;         // lane half
    const int m0   = blockIdx.y * 128;
    const int n0   = blockIdx.x * 128;

    v8f cacc[2][4];
#pragma unroll
    for (int i = 0; i < 2; ++i)
#pragma unroll
        for (int j = 0; j < 4; ++j) cacc[i][j] = zero8();

    // Stage one K-step (32) of A and B into LDS buffer `bufn`.
    auto stage = [&](int bufn, int kbase) {
        if (A_IS_F32) {
            // 128x32 f32 -> bf16, vectorized float4 -> 2x packed bf16 stores.
#pragma unroll
            for (int i = 0; i < 4; ++i) {
                int idx = i * 256 + tid;          // quad index, 1024 total
                int row = idx >> 3, qc = idx & 7; // 8 quads per 32-elem row
                int gr  = m0 + row;
                float4 fv = make_float4(0.f, 0.f, 0.f, 0.f);
                if (gr < M)
                    fv = *(const float4*)((const float*)Aptr +
                                          (size_t)gr * K + kbase + qc * 4);
                uint32_t p0 = (uint32_t)f32_to_bf16(fv.x) |
                              ((uint32_t)f32_to_bf16(fv.y) << 16);
                uint32_t p1 = (uint32_t)f32_to_bf16(fv.z) |
                              ((uint32_t)f32_to_bf16(fv.w) << 16);
                uint2 pk; pk.x = p0; pk.y = p1;
                *(uint2*)&As[bufn][row][qc * 4] = pk;
            }
        } else {
            // 128x32 bf16: async DMA, 16B chunks (4 per row).
#pragma unroll
            for (int i = 0; i < 2; ++i) {
                int idx = i * 256 + tid;          // chunk index, 512 total
                int row = idx >> 2, ch = idx & 3;
                int gr  = m0 + row;
                if (gr < M)
                    async_cp16((const uint16_t*)Aptr +
                                   (size_t)gr * K + kbase + ch * 8,
                               &As[bufn][row][ch * 8]);
            }
        }
        // B tile 128x32 bf16 from transposed weights: pure async copy.
#pragma unroll
        for (int i = 0; i < 2; ++i) {
            int idx = i * 256 + tid;
            int row = idx >> 2, ch = idx & 3;
            async_cp16(Bt + (size_t)(n0 + row) * K + kbase + ch * 8,
                       &Bs[bufn][row][ch * 8]);
        }
    };

    int buf = 0;
    stage(0, 0);

    for (int kb = 0; kb < K; kb += 32) {
        wait_async_lds();
        __syncthreads();
        if (kb + 32 < K) stage(buf ^ 1, kb + 32);

        // A fragments: ISA 16-bit A 16x32 layout
        FragAB afr[2], bfr[4];
#pragma unroll
        for (int ti = 0; ti < 2; ++ti) {
            int row = wm * 32 + ti * 16 + m;
#pragma unroll
            for (int v = 0; v < 8; ++v) {
                int kc = (v < 4) ? (2 * v + 8 * hf) : (16 + 2 * (v - 4) + 8 * hf);
                afr[ti].u[v] = *(const uint32_t*)&As[buf][row][kc];
            }
        }
        // B fragments: same K order as A (lanes0-15 K=0..7,16..23; lanes16-31 K=8..15,24..31)
#pragma unroll
        for (int tj = 0; tj < 4; ++tj) {
            int col = wn * 64 + tj * 16 + m;
#pragma unroll
            for (int v = 0; v < 8; ++v) {
                int kc = (v < 4) ? (2 * v + 8 * hf) : (16 + 2 * (v - 4) + 8 * hf);
                bfr[tj].u[v] = *(const uint32_t*)&Bs[buf][col][kc];
            }
        }
#pragma unroll
        for (int ti = 0; ti < 2; ++ti)
#pragma unroll
            for (int tj = 0; tj < 4; ++tj)
                cacc[ti][tj] = __builtin_amdgcn_wmma_f32_16x16x32_bf16(
                    false, afr[ti].v, false, bfr[tj].v,
                    (short)0, cacc[ti][tj], false, false);
        buf ^= 1;
    }

    // Store: C/D layout -> VGPR r holds rows r (lanes0-15) / r+8 (lanes16-31)
#pragma unroll
    for (int ti = 0; ti < 2; ++ti)
#pragma unroll
        for (int tj = 0; tj < 4; ++tj)
#pragma unroll
            for (int r = 0; r < 8; ++r) {
                int row = m0 + wm * 32 + ti * 16 + r + 8 * hf;
                int col = n0 + wn * 64 + tj * 16 + m;
                if (row < M) {
                    float vv = cacc[ti][tj][r];
                    if (OUT_BF16) {
                        ((uint16_t*)Cptr)[(size_t)row * N + col] = f32_to_bf16(vv);
                    } else {
                        float bs = bias ? bias_scale * bias[col] : 0.0f;
                        ((float*)Cptr)[(size_t)row * N + col] = vv + bs;
                    }
                }
            }
}

// ---------------------------------------------------------------------------
// Fused 3-branch attention: one block = (b*h, 128 rows), 8 waves x 16 query
// rows; WMMA scores, shfl softmax, WMMA P@V, branch-summed.
// ---------------------------------------------------------------------------
#define ATT_S 4096
#define ATT_C 1280
#define ATT_H 20
#define ATT_D 64
#define PSTR  104

template<int SK, int NT, int NKS>
__device__ __forceinline__ void attn_branch(
    const uint16_t* __restrict__ Kp, const uint16_t* __restrict__ Vp,
    int b, int h, int lane, uint16_t* __restrict__ Prow,
    const FragAB (&aq)[2], v8f (&oacc)[4])
{
    const int m  = lane & 15;
    const int hf = lane >> 4;

    for (int idx = lane; idx < 16 * 96; idx += 32)
        Prow[(idx / 96) * PSTR + (idx % 96)] = 0;

    v8f sacc[NT];
#pragma unroll
    for (int t = 0; t < NT; ++t) {
        v8f c = zero8();
        int  key   = t * 16 + m;
        bool valid = key < SK;
        const uint32_t* krow = (const uint32_t*)
            (Kp + (((size_t)b * SK + (valid ? key : 0)) * ATT_C + h * ATT_D));
#pragma unroll
        for (int ks = 0; ks < 2; ++ks) {
            FragAB bk;
#pragma unroll
            for (int v = 0; v < 8; ++v) {
                int kk = ks * 32 + ((v < 4) ? (2 * v + 8 * hf) : (16 + 2 * (v - 4) + 8 * hf));
                bk.u[v] = valid ? krow[kk >> 1] : 0u;
            }
            c = __builtin_amdgcn_wmma_f32_16x16x32_bf16(
                false, aq[ks].v, false, bk.v, (short)0, c, false, false);
        }
        sacc[t] = c;
    }

    const float scale = 0.125f;   // 1/sqrt(64)
#pragma unroll
    for (int r = 0; r < 8; ++r) {
        float tv[NT];
        float mx = -3.0e38f;
#pragma unroll
        for (int t = 0; t < NT; ++t) {
            int col = t * 16 + m;
            float v = sacc[t][r] * scale;
            tv[t] = (col < SK) ? v : -3.0e38f;
            mx = fmaxf(mx, tv[t]);
        }
#pragma unroll
        for (int sh = 1; sh <= 8; sh <<= 1)
            mx = fmaxf(mx, __shfl_xor(mx, sh, 32));
        float pv[NT];
        float sum = 0.0f;
#pragma unroll
        for (int t = 0; t < NT; ++t) {
            float p = __expf(tv[t] - mx);
            p = (t * 16 + m < SK) ? p : 0.0f;
            pv[t] = p;
            sum += p;
        }
#pragma unroll
        for (int sh = 1; sh <= 8; sh <<= 1)
            sum += __shfl_xor(sum, sh, 32);
        float inv = 1.0f / sum;
#pragma unroll
        for (int t = 0; t < NT; ++t)
            Prow[(r + 8 * hf) * PSTR + t * 16 + m] = f32_to_bf16(pv[t] * inv);
    }

#pragma unroll
    for (int ks = 0; ks < NKS; ++ks) {
        FragAB ap;
#pragma unroll
        for (int v = 0; v < 8; ++v) {
            int kc = ks * 32 + ((v < 4) ? (2 * v + 8 * hf)
                                        : (16 + 2 * (v - 4) + 8 * hf));
            ap.u[v] = *(const uint32_t*)&Prow[m * PSTR + kc];
        }
#pragma unroll
        for (int n = 0; n < 4; ++n) {
            FragAB bv;
#pragma unroll
            for (int v = 0; v < 8; ++v) {
                int key  = ks * 32 + ((v < 4) ? (2 * v + 8 * hf) : (16 + 2 * (v - 4) + 8 * hf));
                int dcol = n * 16 + m;
                uint32_t lo = (key < SK)
                    ? (uint32_t)Vp[((size_t)b * SK + key) * ATT_C + h * ATT_D + dcol] : 0u;
                uint32_t hi = (key + 1 < SK)
                    ? (uint32_t)Vp[((size_t)b * SK + key + 1) * ATT_C + h * ATT_D + dcol] : 0u;
                bv.u[v] = lo | (hi << 16);
            }
            oacc[n] = __builtin_amdgcn_wmma_f32_16x16x32_bf16(
                false, ap.v, false, bv.v, (short)0, oacc[n], false, false);
        }
    }
}

__global__ __launch_bounds__(256) void attn_kernel(
    const uint16_t* __restrict__ q,
    const uint16_t* __restrict__ kt,  const uint16_t* __restrict__ vt,
    const uint16_t* __restrict__ kid, const uint16_t* __restrict__ vid,
    const uint16_t* __restrict__ kh,  const uint16_t* __restrict__ vh,
    uint16_t* __restrict__ attn)
{
    __shared__ uint16_t Plds[8 * 16 * PSTR];

    const int lane = threadIdx.x & 31;
    const int w    = threadIdx.x >> 5;
    const int bh   = blockIdx.x;
    const int b    = bh / ATT_H;
    const int h    = bh % ATT_H;
    const int s0   = blockIdx.y * 128 + w * 16;
    const int m    = lane & 15;
    const int hf   = lane >> 4;

    FragAB aq[2];
    const uint32_t* qrow = (const uint32_t*)
        (q + (((size_t)b * ATT_S + s0 + m) * ATT_C + h * ATT_D));
#pragma unroll
    for (int ks = 0; ks < 2; ++ks)
#pragma unroll
        for (int v = 0; v < 8; ++v) {
            int kc = ks * 32 + ((v < 4) ? (2 * v + 8 * hf)
                                        : (16 + 2 * (v - 4) + 8 * hf));
            aq[ks].u[v] = qrow[kc >> 1];
        }

    v8f oacc[4];
#pragma unroll
    for (int n = 0; n < 4; ++n) oacc[n] = zero8();

    uint16_t* Prow = &Plds[w * 16 * PSTR];
    attn_branch<77, 5, 3>(kt,  vt,  b, h, lane, Prow, aq, oacc);  // text
    attn_branch<16, 1, 1>(kid, vid, b, h, lane, Prow, aq, oacc);  // id
    attn_branch<16, 1, 1>(kh,  vh,  b, h, lane, Prow, aq, oacc);  // hair

#pragma unroll
    for (int n = 0; n < 4; ++n)
#pragma unroll
        for (int r = 0; r < 8; ++r) {
            int row = s0 + r + 8 * hf;
            size_t off = ((size_t)b * ATT_S + row) * ATT_C + h * ATT_D + n * 16 + m;
            attn[off] = f32_to_bf16(oacc[n][r]);
        }
}

// ---------------------------------------------------------------------------
// Launch
// ---------------------------------------------------------------------------
extern "C" void kernel_launch(void* const* d_in, const int* in_sizes, int n_in,
                              void* d_out, int out_size, void* d_ws, size_t ws_size,
                              hipStream_t stream)
{
    (void)in_sizes; (void)n_in; (void)out_size; (void)ws_size;

    const float* hidden = (const float*)d_in[0];
    const float* text   = (const float*)d_in[1];
    const float* ids    = (const float*)d_in[2];
    const float* hair   = (const float*)d_in[3];
    const float* Wq     = (const float*)d_in[4];
    const float* Wk     = (const float*)d_in[5];
    const float* Wv     = (const float*)d_in[6];
    const float* Wo     = (const float*)d_in[7];
    const float* bo     = (const float*)d_in[8];
    const float* Wk_id  = (const float*)d_in[9];
    const float* Wv_id  = (const float*)d_in[10];
    const float* Wk_h   = (const float*)d_in[11];
    const float* Wv_h   = (const float*)d_in[12];
    float* out = (float*)d_out;

    char* ws = (char*)d_ws;
    size_t off = 0;
    auto take = [&](size_t bytes) -> char* {
        char* p = ws + off;
        off = (off + bytes + 255) & ~(size_t)255;
        return p;
    };
    uint16_t* qb    = (uint16_t*)take((size_t)8 * 4096 * 1280 * 2);
    uint16_t* ktb   = (uint16_t*)take((size_t)8 * 77 * 1280 * 2);
    uint16_t* vtb   = (uint16_t*)take((size_t)8 * 77 * 1280 * 2);
    uint16_t* kidb  = (uint16_t*)take((size_t)8 * 16 * 1280 * 2);
    uint16_t* vidb  = (uint16_t*)take((size_t)8 * 16 * 1280 * 2);
    uint16_t* khb   = (uint16_t*)take((size_t)8 * 16 * 1280 * 2);
    uint16_t* vhb   = (uint16_t*)take((size_t)8 * 16 * 1280 * 2);
    uint16_t* attnb = (uint16_t*)take((size_t)8 * 4096 * 1280 * 2);
    // transposed bf16 weights [N,K]
    uint16_t* WqT   = (uint16_t*)take((size_t)1280 * 1280 * 2);
    uint16_t* WoT   = (uint16_t*)take((size_t)1280 * 1280 * 2);
    uint16_t* WkT   = (uint16_t*)take((size_t)1280 * 2048 * 2);
    uint16_t* WvT   = (uint16_t*)take((size_t)1280 * 2048 * 2);
    uint16_t* WkiT  = (uint16_t*)take((size_t)1280 * 2048 * 2);
    uint16_t* WviT  = (uint16_t*)take((size_t)1280 * 2048 * 2);
    uint16_t* WkhT  = (uint16_t*)take((size_t)1280 * 2048 * 2);
    uint16_t* WvhT  = (uint16_t*)take((size_t)1280 * 2048 * 2);

    dim3 blk(256);

    // Weight transpose+convert: grid (N/32, K/32)
    transpose_bf16<<<dim3(40, 40), blk, 0, stream>>>(Wq,    WqT,  1280, 1280);
    transpose_bf16<<<dim3(40, 40), blk, 0, stream>>>(Wo,    WoT,  1280, 1280);
    transpose_bf16<<<dim3(40, 64), blk, 0, stream>>>(Wk,    WkT,  2048, 1280);
    transpose_bf16<<<dim3(40, 64), blk, 0, stream>>>(Wv,    WvT,  2048, 1280);
    transpose_bf16<<<dim3(40, 64), blk, 0, stream>>>(Wk_id, WkiT, 2048, 1280);
    transpose_bf16<<<dim3(40, 64), blk, 0, stream>>>(Wv_id, WviT, 2048, 1280);
    transpose_bf16<<<dim3(40, 64), blk, 0, stream>>>(Wk_h,  WkhT, 2048, 1280);
    transpose_bf16<<<dim3(40, 64), blk, 0, stream>>>(Wv_h,  WvhT, 2048, 1280);

    // q = hidden @ Wq   -> bf16
    gemm_wmma<true, true><<<dim3(10, 256), blk, 0, stream>>>(
        hidden, WqT, qb, nullptr, 0.0f, 32768, 1280, 1280);
    // K/V projections -> bf16
    gemm_wmma<true, true><<<dim3(10, 5), blk, 0, stream>>>(
        text, WkT, ktb, nullptr, 0.0f, 616, 1280, 2048);
    gemm_wmma<true, true><<<dim3(10, 5), blk, 0, stream>>>(
        text, WvT, vtb, nullptr, 0.0f, 616, 1280, 2048);
    gemm_wmma<true, true><<<dim3(10, 1), blk, 0, stream>>>(
        ids, WkiT, kidb, nullptr, 0.0f, 128, 1280, 2048);
    gemm_wmma<true, true><<<dim3(10, 1), blk, 0, stream>>>(
        ids, WviT, vidb, nullptr, 0.0f, 128, 1280, 2048);
    gemm_wmma<true, true><<<dim3(10, 1), blk, 0, stream>>>(
        hair, WkhT, khb, nullptr, 0.0f, 128, 1280, 2048);
    gemm_wmma<true, true><<<dim3(10, 1), blk, 0, stream>>>(
        hair, WvhT, vhb, nullptr, 0.0f, 128, 1280, 2048);

    // fused 3-branch attention -> summed bf16 context
    attn_kernel<<<dim3(8 * 20, 4096 / 128), blk, 0, stream>>>(
        qb, ktb, vtb, kidb, vidb, khb, vhb, attnb);

    // out = (sum of branch contexts) @ Wo + 3*bo  -> f32  (A is bf16: async)
    gemm_wmma<false, false><<<dim3(10, 256), blk, 0, stream>>>(
        attnb, WoT, out, bo, 3.0f, 32768, 1280, 1280);
}
